// TemporalGATBlock_76725295775918
// MI455X (gfx1250) — matrix operand order repaired and checked by reading, not compiled, not run on hardware
//
#include <hip/hip_runtime.h>
#include <hip/hip_bf16.h>

typedef __attribute__((ext_vector_type(4)))  float    v4f;
typedef __attribute__((ext_vector_type(8)))  float    v8f;
typedef __attribute__((ext_vector_type(4)))  _Float16 v4h;
typedef __attribute__((ext_vector_type(8)))  _Float16 v8h;
typedef __attribute__((ext_vector_type(16))) _Float16 v16h;

#define HC 128           // HEADS*OUT_CH == IN_CH == 128
#define HEADS 4
#define NEG_SLOPE 0.2f
#define LN_EPS 1e-5f
#define TPAD 132         // 16x132 LDS tile rows (pad kills half-wave bank aliasing)

// ---------------- convert x to f16 ----------------
__global__ __launch_bounds__(256) void gat_cvt_x(const float* __restrict__ x,
                                                 _Float16* __restrict__ xh, int n4) {
    int i = blockIdx.x * blockDim.x + threadIdx.x;
    if (i >= n4) return;
    v4f v = *(const v4f*)(x + (size_t)i * 4);
    v4h h = {(_Float16)v.x, (_Float16)v.y, (_Float16)v.z, (_Float16)v.w};
    *(v4h*)(xh + (size_t)i * 4) = h;
}

// ------ pack W_src/W_dst into WMMA B-matrix lane layout (f16) ------
// wpack[((mat*4+kc)*8 + ntile)*32 + lane][0..15] = W[kc*32 + (lane>=16?8:0) + e + (e>=8?8:0)][ntile*16 + (lane&15)]
__global__ __launch_bounds__(256) void gat_pack_w(const float* __restrict__ Wsrc,
                                                  const float* __restrict__ Wdst,
                                                  _Float16* __restrict__ wpack) {
    int t = blockIdx.x * blockDim.x + threadIdx.x;   // 0..2047
    if (t >= 2 * 4 * 8 * 32) return;
    int lane = t & 31, rest = t >> 5;
    int ntile = rest & 7; rest >>= 3;
    int kc = rest & 3;  int mat = rest >> 2;
    const float* W = mat ? Wdst : Wsrc;
    int kbase = kc * 32 + ((lane & 16) ? 8 : 0);
    int col = ntile * 16 + (lane & 15);
    _Float16* op = wpack + ((size_t)t << 4);
    #pragma unroll
    for (int e = 0; e < 16; ++e)
        op[e] = (_Float16)W[(size_t)(kbase + e + (e >= 8 ? 8 : 0)) * HC + col];
}

// ---------------- init accum / emax / denom ----------------
__global__ __launch_bounds__(256) void gat_init(float* __restrict__ out,
                                                float* __restrict__ emax,
                                                float* __restrict__ denom, int N) {
    int i = blockIdx.x * blockDim.x + threadIdx.x;
    int total = N * HC;
    if (i < total) out[i] = 0.0f;
    if (i < N * HEADS) { emax[i] = -__builtin_inff(); denom[i] = 0.0f; }
}

// ---------------- WMMA GEMM: hs = x@Wsrc, hd = x@Wdst ----------------
// Block = 16 rows; 8 waves = 8 column tiles of 16; each wave computes both matrices.
// Epilogue stages tiles in LDS, then emits fully coalesced 512B row stores.
__global__ __launch_bounds__(256) void gat_gemm_wmma(const _Float16* __restrict__ xh,
                                                     const _Float16* __restrict__ wpack,
                                                     float* __restrict__ hs,
                                                     float* __restrict__ hd, int N) {
    __shared__ float tileS[16 * TPAD];
    __shared__ float tileD[16 * TPAD];

    const int lane = threadIdx.x & 31;
    const int wave = threadIdx.x >> 5;          // ntile 0..7 (16 cols each)
    const int blockRow = blockIdx.x * 16;
    int arow = blockRow + (lane & 15);
    if (arow >= N) arow = N - 1;                // clamp (stores are guarded)
    const int halfSel = (lane & 16) ? 8 : 0;    // A frag: lanes16-31 hold K=8..15,24..31

    v16h a[4];
    const _Float16* rp = xh + (size_t)arow * HC;
    #pragma unroll
    for (int kc = 0; kc < 4; ++kc) {
        v8h lo = *(const v8h*)(rp + kc * 32 + halfSel);
        v8h hi = *(const v8h*)(rp + kc * 32 + halfSel + 16);
        a[kc] = __builtin_shufflevector(lo, hi, 0,1,2,3,4,5,6,7,8,9,10,11,12,13,14,15);
    }

    const int col = wave * 16 + (lane & 15);
    const int mbase = (lane & 16) ? 8 : 0;

    #pragma unroll
    for (int mat = 0; mat < 2; ++mat) {
        v8f acc = {};
        #pragma unroll
        for (int kc = 0; kc < 4; ++kc) {
            const _Float16* bp = wpack + ((((size_t)(mat * 4 + kc) * 8 + wave) * 32 + lane) << 4);
            v8h lo = *(const v8h*)(bp);
            v8h hi = *(const v8h*)(bp + 8);
            v16h b = __builtin_shufflevector(lo, hi, 0,1,2,3,4,5,6,7,8,9,10,11,12,13,14,15);
            acc = __builtin_amdgcn_wmma_f32_16x16x32_f16(false, a[kc], false, b,
                                                         (short)0, acc, false, false);
        }
        float* tl = mat ? tileD : tileS;
        #pragma unroll
        for (int r = 0; r < 8; ++r)
            tl[(mbase + r) * TPAD + col] = acc[r];
    }

    __syncthreads();

    // coalesced write-out: 16 rows x 128 cols per matrix = 512 v4f chunks
    #pragma unroll
    for (int c = threadIdx.x; c < 512; c += 256) {
        int row = c >> 5;             // 0..15
        int cb  = (c & 31) * 4;       // column base
        int grow = blockRow + row;
        if (grow < N) {
            v4f vs = *(const v4f*)(tileS + row * TPAD + cb);
            v4f vd = *(const v4f*)(tileD + row * TPAD + cb);
            *(v4f*)(hs + (size_t)grow * HC + cb) = vs;
            *(v4f*)(hd + (size_t)grow * HC + cb) = vd;
        }
    }
}

// ---------------- float atomic max (int-punning trick) ----------------
__device__ __forceinline__ void atomicMaxF32(float* addr, float v) {
    if (v >= 0.0f) atomicMax((int*)addr, __float_as_int(v));
    else           atomicMin((unsigned int*)addr, __float_as_uint(v));
}

// ---------------- pass 1: per-edge attention scores + segment max ----------------
__global__ __launch_bounds__(256) void gat_edge_scores(const long long* __restrict__ ei,
                                                       const float* __restrict__ hs,
                                                       const float* __restrict__ hd,
                                                       const float* __restrict__ att,
                                                       float* __restrict__ ebuf,
                                                       float* __restrict__ emax,
                                                       int NE, int NV) {
    const int lane = threadIdx.x & 31;
    long long edge = (long long)blockIdx.x * 8 + (threadIdx.x >> 5);
    long long ET = (long long)NE + NV;
    if (edge >= ET) return;
    int s, d;
    if (edge < NE) { s = (int)ei[edge]; d = (int)ei[NE + edge]; }
    else           { s = d = (int)(edge - NE); }

    v4f a = *(const v4f*)(hs + (size_t)s * HC + lane * 4);
    v4f b = *(const v4f*)(hd + (size_t)d * HC + lane * 4);
    v4f w = *(const v4f*)(att + lane * 4);
    float p = 0.0f, m;
    m = a.x + b.x; m = (m > 0.0f) ? m : NEG_SLOPE * m; p += m * w.x;
    m = a.y + b.y; m = (m > 0.0f) ? m : NEG_SLOPE * m; p += m * w.y;
    m = a.z + b.z; m = (m > 0.0f) ? m : NEG_SLOPE * m; p += m * w.z;
    m = a.w + b.w; m = (m > 0.0f) ? m : NEG_SLOPE * m; p += m * w.w;

    p += __shfl_xor(p, 1, 32);
    p += __shfl_xor(p, 2, 32);
    p += __shfl_xor(p, 4, 32);          // 8-lane group = one head

    if ((lane & 7) == 0) {
        int h = lane >> 3;
        ebuf[edge * HEADS + h] = p;
        atomicMaxF32(emax + (size_t)d * HEADS + h, p);
    }
}

// ---------------- pass 2: exp(e - max) + segment sum ----------------
__global__ __launch_bounds__(256) void gat_edge_exp(const long long* __restrict__ ei,
                                                    float* __restrict__ ebuf,
                                                    const float* __restrict__ emax,
                                                    float* __restrict__ denom,
                                                    int NE, int NV) {
    long long i = (long long)blockIdx.x * blockDim.x + threadIdx.x;
    long long ET4 = ((long long)NE + NV) * HEADS;
    if (i >= ET4) return;
    long long edge = i >> 2; int h = (int)(i & 3);
    int d = (edge < NE) ? (int)ei[NE + edge] : (int)(edge - NE);
    float ee = __expf(ebuf[i] - emax[(size_t)d * HEADS + h]);
    ebuf[i] = ee;
    atomicAdd(denom + (size_t)d * HEADS + h, ee);
}

// ---------------- pass 3: weighted scatter-aggregate ----------------
__global__ __launch_bounds__(256) void gat_edge_aggr(const long long* __restrict__ ei,
                                                     const float* __restrict__ hs,
                                                     const float* __restrict__ ebuf,
                                                     const float* __restrict__ denom,
                                                     float* __restrict__ out,
                                                     int NE, int NV) {
    const int lane = threadIdx.x & 31;
    long long edge = (long long)blockIdx.x * 8 + (threadIdx.x >> 5);
    long long ET = (long long)NE + NV;
    if (edge >= ET) return;
    int s, d;
    if (edge < NE) { s = (int)ei[edge]; d = (int)ei[NE + edge]; }
    else           { s = d = (int)(edge - NE); }
    int h = lane >> 3;
    float alpha = ebuf[edge * HEADS + h] / denom[(size_t)d * HEADS + h];
    v4f a = *(const v4f*)(hs + (size_t)s * HC + lane * 4);
    float* op = out + (size_t)d * HC + lane * 4;
    atomicAdd(op + 0, alpha * a.x);
    atomicAdd(op + 1, alpha * a.y);
    atomicAdd(op + 2, alpha * a.z);
    atomicAdd(op + 3, alpha * a.w);
}

// ---------------- bias + LayerNorm + ELU (one wave32 per node) ----------------
__global__ __launch_bounds__(256) void gat_ln_elu(float* __restrict__ out,
                                                  const float* __restrict__ bias,
                                                  const float* __restrict__ gamma,
                                                  const float* __restrict__ beta,
                                                  int NV) {
    const int lane = threadIdx.x & 31;
    int node = blockIdx.x * 8 + (threadIdx.x >> 5);
    if (node >= NV) return;
    float* row = out + (size_t)node * HC;
    v4f v  = *(v4f*)(row + lane * 4);
    v4f bb = *(const v4f*)(bias + lane * 4);
    v.x += bb.x; v.y += bb.y; v.z += bb.z; v.w += bb.w;

    float sum = v.x + v.y + v.z + v.w;
    #pragma unroll
    for (int m = 1; m < 32; m <<= 1) sum += __shfl_xor(sum, m, 32);
    float mean = sum * (1.0f / HC);

    float dx = v.x - mean, dy = v.y - mean, dz = v.z - mean, dw = v.w - mean;
    float s2 = dx * dx + dy * dy + dz * dz + dw * dw;
    #pragma unroll
    for (int m = 1; m < 32; m <<= 1) s2 += __shfl_xor(s2, m, 32);
    float inv = rsqrtf(s2 * (1.0f / HC) + LN_EPS);

    v4f g = *(const v4f*)(gamma + lane * 4);
    v4f b2 = *(const v4f*)(beta + lane * 4);
    float o;
    o = dx * inv * g.x + b2.x; v.x = (o > 0.0f) ? o : (__expf(o) - 1.0f);
    o = dy * inv * g.y + b2.y; v.y = (o > 0.0f) ? o : (__expf(o) - 1.0f);
    o = dz * inv * g.z + b2.z; v.z = (o > 0.0f) ? o : (__expf(o) - 1.0f);
    o = dw * inv * g.w + b2.w; v.w = (o > 0.0f) ? o : (__expf(o) - 1.0f);
    *(v4f*)(row + lane * 4) = v;
}

extern "C" void kernel_launch(void* const* d_in, const int* in_sizes, int n_in,
                              void* d_out, int out_size, void* d_ws, size_t ws_size,
                              hipStream_t stream) {
    const float*     x     = (const float*)d_in[0];
    const long long* ei    = (const long long*)d_in[1];
    const float*     Wsrc  = (const float*)d_in[2];
    const float*     Wdst  = (const float*)d_in[3];
    const float*     att   = (const float*)d_in[4];
    const float*     bias  = (const float*)d_in[5];
    const float*     gamma = (const float*)d_in[6];
    const float*     beta  = (const float*)d_in[7];
    float* out = (float*)d_out;

    const int N  = in_sizes[0] / HC;      // 50000
    const int E  = in_sizes[1] / 2;       // 1,600,000
    const int ET = E + N;                 // edges + self loops

    // workspace carve (poisoned each run; everything re-initialized below)
    char* w = (char*)d_ws;
    auto carve = [&](size_t bytes) { void* p = (void*)w; w += (bytes + 255) & ~(size_t)255; return p; };
    _Float16* xh    = (_Float16*)carve((size_t)N * HC * 2);
    _Float16* wpack = (_Float16*)carve((size_t)2 * HC * HC * 2);
    float*    hs    = (float*)carve((size_t)N * HC * 4);
    float*    hd    = (float*)carve((size_t)N * HC * 4);
    float*    ebuf  = (float*)carve((size_t)ET * HEADS * 4);
    float*    emax  = (float*)carve((size_t)N * HEADS * 4);
    float*    denom = (float*)carve((size_t)N * HEADS * 4);

    int n4 = N * (HC / 4);
    gat_cvt_x<<<(n4 + 255) / 256, 256, 0, stream>>>(x, xh, n4);
    gat_pack_w<<<8, 256, 0, stream>>>(Wsrc, Wdst, wpack);
    gat_init<<<(N * HC + 255) / 256, 256, 0, stream>>>(out, emax, denom, N);
    gat_gemm_wmma<<<(N + 15) / 16, 256, 0, stream>>>(xh, wpack, hs, hd, N);
    gat_edge_scores<<<(ET + 7) / 8, 256, 0, stream>>>(ei, hs, hd, att, ebuf, emax, E, N);
    gat_edge_exp<<<(ET * HEADS + 255) / 256, 256, 0, stream>>>(ei, ebuf, emax, denom, E, N);
    gat_edge_aggr<<<(ET + 7) / 8, 256, 0, stream>>>(ei, hs, ebuf, denom, out, E, N);
    gat_ln_elu<<<(N + 7) / 8, 256, 0, stream>>>(out, bias, gamma, beta, N);
}
